// OptimizedMultiAttentionHead_88476326298316
// MI455X (gfx1250) — matrix operand order repaired and checked by reading, not compiled, not run on hardware
//
#include <hip/hip_runtime.h>
#include <hip/hip_bf16.h>

// ---------------------------------------------------------------------------
// MI455X (gfx1250) fused causal multi-head attention.
// bf16 WMMA pipeline + Tensor Data Mover (TDM) LDS staging for GEMM B-slabs.
// ---------------------------------------------------------------------------

typedef __bf16 bf16_t;
typedef bf16_t v16bf __attribute__((ext_vector_type(16)));
typedef bf16_t v8bf  __attribute__((ext_vector_type(8)));
typedef float  v8f   __attribute__((ext_vector_type(8)));
typedef unsigned int u32x4 __attribute__((ext_vector_type(4)));
typedef int i32x8 __attribute__((ext_vector_type(8)));
typedef int i32x4 __attribute__((ext_vector_type(4)));

union Frag { v16bf v; v8bf h[2]; };

static constexpr int Bc = 4;
static constexpr int Tc = 1024;
static constexpr int Dc = 1024;
static constexpr int Hc = 16;
static constexpr int Sc = 64;

// --- WMMA wrapper: D = A(16x32 bf16) x B(32x16 bf16) + C(16x16 f32) --------
__device__ __forceinline__ v8f wmma_bf(v16bf a, v16bf b, v8f c) {
  return __builtin_amdgcn_wmma_f32_16x16x32_bf16(
      /*neg_a=*/false, a, /*neg_b=*/false, b,
      /*c_mod=*/(short)0, c, /*reuse_a=*/false, /*reuse_b=*/false);
}

// A-operand (16x32, MxK): lane holds row (lane&15); K pattern per ISA 7.12.2:
//   VGPR0-3: K in [half*8, half*8+8), VGPR4-7: K in [half*8+16, half*8+24)
__device__ __forceinline__ v16bf loadA32(const bf16_t* row, int kbase, int half) {
  Frag f;
  f.h[0] = *(const v8bf*)(row + kbase + half * 8);
  f.h[1] = *(const v8bf*)(row + kbase + half * 8 + 16);
  return f.v;
}

// B-operand (32x16, KxN): lane holds column (lane&15); same K pattern as A:
//   VGPR0-3: K in [half*8, half*8+8), VGPR4-7: K in [half*8+16, half*8+24)
__device__ __forceinline__ v16bf loadB32(const bf16_t* row, int kbase, int half) {
  Frag f;
  f.h[0] = *(const v8bf*)(row + kbase + half * 8);
  f.h[1] = *(const v8bf*)(row + kbase + half * 8 + 16);
  return f.v;
}

// ---------------------------------------------------------------------------
// Tensor Data Mover: 2-D tile (rows x cols bf16) from global -> LDS.
// Descriptor per CDNA5 ISA §8.3 (group0) / §8.4 (group1); groups 2/3 zero
// (2-D tile, no iterate/gather/padding/multicast). 6-arg builtin form
// (clang-23 / therock-10 toolchain).
// ---------------------------------------------------------------------------
__device__ __forceinline__ void tdm_load_2d_bf16(unsigned int lds_byte_addr,
                                                 const void* gaddr,
                                                 unsigned int rows,
                                                 unsigned int cols,
                                                 unsigned int row_stride_elems) {
  const unsigned long long ga = (unsigned long long)(uintptr_t)gaddr;
  u32x4 g0;
  g0[0] = 1u;                                   // count=1, user descriptor
  g0[1] = lds_byte_addr;                        // lds_addr (bytes)
  g0[2] = (unsigned int)(ga & 0xFFFFFFFFu);     // global_addr[31:0]
  g0[3] = (unsigned int)((ga >> 32) & 0x1FFFFFFu) | (2u << 30);  // type=2

  // group1: data_size=1 (2 bytes); huge tensor dims (no OOB clipping);
  // tile_dim0 = cols (contiguous), tile_dim1 = rows; dim0 stride = row stride.
  const unsigned int td0 = 0x7FFFFFFFu, td1 = 0x7FFFFFFFu;
  i32x8 g1;
  g1[0] = (int)(1u << 16);                                      // data_size
  g1[1] = (int)((td0 & 0xFFFFu) << 16);                         // tdim0 lo
  g1[2] = (int)(((td0 >> 16) & 0xFFFFu) | ((td1 & 0xFFFFu) << 16));
  g1[3] = (int)(((td1 >> 16) & 0xFFFFu) | ((cols & 0xFFFFu) << 16));
  g1[4] = (int)(rows & 0xFFFFu);                                // tile_dim1
  g1[5] = (int)row_stride_elems;                                // stride lo32
  g1[6] = 0;                                                    // stride hi
  g1[7] = 0;
  const i32x4 gz  = {0, 0, 0, 0};
  const i32x8 gz8 = {0, 0, 0, 0, 0, 0, 0, 0};
  __builtin_amdgcn_tensor_load_to_lds(g0, g1, gz, gz, gz8, 0);
}

// low 32 bits of a flat LDS pointer = LDS byte offset (aperture mapping)
__device__ __forceinline__ unsigned int lds_off(const void* p) {
  return (unsigned int)(uintptr_t)p;
}

// Cross-lane reductions within each 16-lane group: ds_swizzle XOR 1,2,4,8.
__device__ __forceinline__ float red_max16(float v) {
  v = fmaxf(v, __int_as_float(__builtin_amdgcn_ds_swizzle(__float_as_int(v), 0x041f)));
  v = fmaxf(v, __int_as_float(__builtin_amdgcn_ds_swizzle(__float_as_int(v), 0x081f)));
  v = fmaxf(v, __int_as_float(__builtin_amdgcn_ds_swizzle(__float_as_int(v), 0x101f)));
  v = fmaxf(v, __int_as_float(__builtin_amdgcn_ds_swizzle(__float_as_int(v), 0x201f)));
  return v;
}
__device__ __forceinline__ float red_sum16(float v) {
  v += __int_as_float(__builtin_amdgcn_ds_swizzle(__float_as_int(v), 0x041f));
  v += __int_as_float(__builtin_amdgcn_ds_swizzle(__float_as_int(v), 0x081f));
  v += __int_as_float(__builtin_amdgcn_ds_swizzle(__float_as_int(v), 0x101f));
  v += __int_as_float(__builtin_amdgcn_ds_swizzle(__float_as_int(v), 0x201f));
  return v;
}

// ---------------------------------------------------------------------------
// fp32 -> bf16 elementwise convert
// ---------------------------------------------------------------------------
__global__ void conv_f2bf_kernel(const float* __restrict__ src,
                                 bf16_t* __restrict__ dst, int n) {
  int i = blockIdx.x * 256 + threadIdx.x;
  if (i < n) dst[i] = (bf16_t)src[i];
}

// fp32 (H,D,S) -> bf16 (H,S,D) transpose-convert (QKV weights)
__global__ void conv_wT_kernel(const float* __restrict__ src,
                               bf16_t* __restrict__ dst) {
  int i = blockIdx.x * 256 + threadIdx.x;  // i over H*S*D
  int d = i % Dc;
  int s = (i / Dc) % Sc;
  int h = i / (Dc * Sc);
  dst[i] = (bf16_t)src[((size_t)h * Dc + d) * Sc + s];
}

// ---------------------------------------------------------------------------
// QKV projection. Block = 4 waves sharing one (w,b,h) weight slab (64 x D).
// TDM double-buffers 64x128 weight tiles into LDS; each wave owns one 16-row
// x-tile and produces a 16x64 output tile. Q pre-scaled by 1/sqrt(S);
// V stored transposed (b,h,s,t).
// ---------------------------------------------------------------------------
__global__ __launch_bounds__(128)
void qkv_kernel(const bf16_t* __restrict__ xb,
                const bf16_t* __restrict__ wqT,
                const bf16_t* __restrict__ wkT,
                const bf16_t* __restrict__ wvT,
                bf16_t* __restrict__ Q, bf16_t* __restrict__ K,
                bf16_t* __restrict__ Vt) {
  __shared__ __align__(16) bf16_t wbuf[2][64 * 128];  // 32 KB ping-pong

  const int wid = threadIdx.x >> 5;
  const int TtG = (Tc / 16) / 4;  // t-tile groups per (w,b,h)
  const int bi  = blockIdx.x;
  const int ttg = bi % TtG;
  const int h   = (bi / TtG) % Hc;
  const int b   = (bi / (TtG * Hc)) % Bc;
  const int w   = bi / (TtG * Hc * Bc);  // 0=Q 1=K 2=V
  const int tt  = ttg * 4 + wid;

  const int lane = threadIdx.x & 31;
  const int l16  = lane & 15;
  const int half = lane >> 4;
  const int k8   = half * 8;

  const bf16_t* wT = (w == 0) ? wqT : ((w == 1) ? wkT : wvT);
  const bf16_t* wslab = wT + (size_t)h * Sc * Dc;  // 64 rows x D

  const v8f vzero = {0.f, 0.f, 0.f, 0.f, 0.f, 0.f, 0.f, 0.f};
  v8f o[4] = {vzero, vzero, vzero, vzero};

  const bf16_t* xrow = xb + ((size_t)(b * Tc) + tt * 16 + l16) * Dc;

  constexpr int NCH = Dc / 128;  // 8 macro-chunks of K=128
  if (wid == 0)
    tdm_load_2d_bf16(lds_off(&wbuf[0][0]), wslab, 64, 128, Dc);

  for (int dc = 0; dc < NCH; ++dc) {
    if (wid == 0) {
      if (dc + 1 < NCH) {
        tdm_load_2d_bf16(lds_off(&wbuf[(dc + 1) & 1][0]),
                         wslab + (size_t)(dc + 1) * 128, 64, 128, Dc);
        __builtin_amdgcn_s_wait_tensorcnt(1);  // chunk dc complete
      } else {
        __builtin_amdgcn_s_wait_tensorcnt(0);
      }
    }
    __syncthreads();  // weight tile dc visible to all 4 waves

    if (dc + 1 < NCH)  // prefetch next A rows (global_prefetch_b8)
      __builtin_prefetch(xrow + (dc + 1) * 128, 0, 3);

    Frag ax[4];
#pragma unroll
    for (int kc = 0; kc < 4; ++kc)
      ax[kc].v = loadA32(xrow, dc * 128 + kc * 32, half);

    const bf16_t* lw = &wbuf[dc & 1][0];
#pragma unroll
    for (int kc = 0; kc < 4; ++kc) {
#pragma unroll
      for (int st = 0; st < 4; ++st) {
        const bf16_t* brow = lw + (st * 16 + l16) * 128 + kc * 32;
        o[st] = wmma_bf(ax[kc].v, loadB32(brow, 0, half), o[st]);
      }
    }
    __syncthreads();  // done with tile dc before TDM overwrites this buffer
  }

  const float scale = (w == 0) ? 0.125f : 1.0f;  // 1/sqrt(64)
#pragma unroll
  for (int st = 0; st < 4; ++st) {
#pragma unroll
    for (int r = 0; r < 8; ++r) {
      const int t = tt * 16 + r + k8;
      const int s = st * 16 + l16;
      const bf16_t val = (bf16_t)(o[st][r] * scale);
      if (w == 2)
        Vt[((size_t)(b * Hc + h) * Sc + s) * Tc + t] = val;
      else if (w == 0)
        Q[((size_t)(b * Hc + h) * Tc + t) * Sc + s] = val;
      else
        K[((size_t)(b * Hc + h) * Tc + t) * Sc + s] = val;
    }
  }
}

// ---------------------------------------------------------------------------
// Two-pass flash attention with post-softmax learned (cos) dropout.
// One wave per 16-row query tile; 4 waves of a block cover consecutive query
// tiles of the same (b,h) so K/V tiles are reused through WGP$/L2.
// ---------------------------------------------------------------------------
__global__ __launch_bounds__(128)
void attn_kernel(const bf16_t* __restrict__ Q, const bf16_t* __restrict__ K,
                 const bf16_t* __restrict__ Vt,
                 const float* __restrict__ A1, const float* __restrict__ B1,
                 bf16_t* __restrict__ AO) {
  __shared__ __align__(16) bf16_t pbuf[4][16 * 32];

  const int wid  = threadIdx.x >> 5;
  const int widx = blockIdx.x * 4 + wid;
  const int Tt   = Tc / 16;
  const int qt = widx % Tt;
  const int h  = (widx / Tt) % Hc;
  const int b  = widx / (Tt * Hc);

  const int lane = threadIdx.x & 31;
  const int l16  = lane & 15;
  const int half = lane >> 4;
  const int k8   = half * 8;

  const size_t bh = (size_t)(b * Hc + h);
  const bf16_t* qrow = Q + (bh * Tc + qt * 16 + l16) * Sc;
  const v16bf aq0 = loadA32(qrow, 0, half);
  const v16bf aq1 = loadA32(qrow, 32, half);

  const v8f vzero = {0.f, 0.f, 0.f, 0.f, 0.f, 0.f, 0.f, 0.f};

  // -------- Pass A: row max m and row sum l over all causal keys ----------
  float m[8], l[8];
#pragma unroll
  for (int r = 0; r < 8; ++r) { m[r] = -3.0e38f; l[r] = 0.f; }

  for (int u = 0; u <= qt; ++u) {
    const bf16_t* krow = K + (bh * Tc + u * 16 + l16) * Sc;
    v8f c = wmma_bf(aq0, loadB32(krow, 0, half), vzero);
    c = wmma_bf(aq1, loadB32(krow, 32, half), c);
    const bool diag = (u == qt);
#pragma unroll
    for (int r = 0; r < 8; ++r) {
      const int mrow = r + k8;                 // row within 16-row tile
      float s = c[r];                          // col n = l16
      if (diag && (l16 > mrow)) s = -3.0e38f;  // causal mask
      const float rm = red_max16(s);
      const float mn = fmaxf(m[r], rm);
      const float e  = __expf(s - mn);
      const float rs = red_sum16(e);
      l[r] = l[r] * __expf(m[r] - mn) + rs;
      m[r] = mn;
    }
  }

  float invl[8];
#pragma unroll
  for (int r = 0; r < 8; ++r) invl[r] = 1.0f / l[r];

  // -------- Pass B: P = dropout(softmax(S)); O += P @ V -------------------
  v8f o[4] = {vzero, vzero, vzero, vzero};
  bf16_t* pb = pbuf[wid];

  for (int uc = 0; uc <= (qt >> 1); ++uc) {
#pragma unroll
    for (int du = 0; du < 2; ++du) {
      const int u = uc * 2 + du;
      const int colb = du * 16;
      if (u <= qt) {
        const bf16_t* krow = K + (bh * Tc + u * 16 + l16) * Sc;
        v8f c = wmma_bf(aq0, loadB32(krow, 0, half), vzero);
        c = wmma_bf(aq1, loadB32(krow, 32, half), c);
        const float a1 = A1[u * 16 + l16];
        const float b1 = B1[u * 16 + l16];
        const bool diag = (u == qt);
#pragma unroll
        for (int r = 0; r < 8; ++r) {
          const int mrow = r + k8;
          float p = __expf(c[r] - m[r]) * invl[r];
          if (diag && (l16 > mrow)) p = 0.f;
          const float msk = 0.5f * __cosf(a1 * p + b1) + 0.5f;
          pb[mrow * 32 + colb + l16] = (bf16_t)(p * msk);
        }
      } else {
#pragma unroll
        for (int r = 0; r < 8; ++r)
          pb[(r + k8) * 32 + colb + l16] = (bf16_t)0.f;
      }
    }
    // make per-lane LDS writes visible wave-wide before A-layout reads
    asm volatile("s_wait_dscnt 0" ::: "memory");
    Frag ap;
    ap.h[0] = *(const v8bf*)(pb + l16 * 32 + k8);
    ap.h[1] = *(const v8bf*)(pb + l16 * 32 + k8 + 16);

    const int u0 = uc * 32;
#pragma unroll
    for (int st = 0; st < 4; ++st) {
      const bf16_t* vrow = Vt + (bh * Sc + st * 16 + l16) * Tc + u0;
      o[st] = wmma_bf(ap.v, loadB32(vrow, 0, half), o[st]);
    }
  }

  // store O tile as bf16, layout (b, t, h, s) == (b, t, D) after flatten
#pragma unroll
  for (int st = 0; st < 4; ++st) {
#pragma unroll
    for (int r = 0; r < 8; ++r) {
      const int t = qt * 16 + r + k8;
      const int s = st * 16 + l16;
      AO[((size_t)(b * Tc + t) * Hc + h) * Sc + s] = (bf16_t)o[st][r];
    }
  }
}

// ---------------------------------------------------------------------------
// Output projection: out = AO @ proj_w^T + b, then cos-dropout, fp32 out.
// Block = 4 waves sharing one 64-row proj_w slab (TDM-staged), each wave a
// 16x64 output tile.
// ---------------------------------------------------------------------------
__global__ __launch_bounds__(128)
void proj_kernel(const bf16_t* __restrict__ AO, const bf16_t* __restrict__ pwb,
                 const float* __restrict__ bias,
                 const float* __restrict__ A2, const float* __restrict__ B2,
                 float* __restrict__ out) {
  __shared__ __align__(16) bf16_t wbuf[2][64 * 128];  // 32 KB ping-pong

  const int wid = threadIdx.x >> 5;
  const int bi  = blockIdx.x;
  const int jt  = bi % (Dc / 64);
  const int itg = bi / (Dc / 64);
  const int it  = itg * 4 + wid;

  const int lane = threadIdx.x & 31;
  const int l16  = lane & 15;
  const int half = lane >> 4;
  const int k8   = half * 8;

  const bf16_t* wslab = pwb + (size_t)(jt * 64) * Dc;  // 64 rows x D

  const v8f vzero = {0.f, 0.f, 0.f, 0.f, 0.f, 0.f, 0.f, 0.f};
  v8f o[4] = {vzero, vzero, vzero, vzero};

  const bf16_t* arow = AO + ((size_t)it * 16 + l16) * Dc;

  constexpr int NCH = Dc / 128;
  if (wid == 0)
    tdm_load_2d_bf16(lds_off(&wbuf[0][0]), wslab, 64, 128, Dc);

  for (int dc = 0; dc < NCH; ++dc) {
    if (wid == 0) {
      if (dc + 1 < NCH) {
        tdm_load_2d_bf16(lds_off(&wbuf[(dc + 1) & 1][0]),
                         wslab + (size_t)(dc + 1) * 128, 64, 128, Dc);
        __builtin_amdgcn_s_wait_tensorcnt(1);
      } else {
        __builtin_amdgcn_s_wait_tensorcnt(0);
      }
    }
    __syncthreads();

    if (dc + 1 < NCH)
      __builtin_prefetch(arow + (dc + 1) * 128, 0, 3);

    Frag ax[4];
#pragma unroll
    for (int kc = 0; kc < 4; ++kc)
      ax[kc].v = loadA32(arow, dc * 128 + kc * 32, half);

    const bf16_t* lw = &wbuf[dc & 1][0];
#pragma unroll
    for (int kc = 0; kc < 4; ++kc) {
#pragma unroll
      for (int st = 0; st < 4; ++st) {
        const bf16_t* brow = lw + (st * 16 + l16) * 128 + kc * 32;
        o[st] = wmma_bf(ax[kc].v, loadB32(brow, 0, half), o[st]);
      }
    }
    __syncthreads();
  }

#pragma unroll
  for (int st = 0; st < 4; ++st) {
    const int j = jt * 64 + st * 16 + l16;
    const float bj = bias[j];
    const float a2 = A2[j];
    const float b2 = B2[j];
#pragma unroll
    for (int r = 0; r < 8; ++r) {
      const int i = it * 16 + r + k8;
      float v = o[st][r] + bj;
      v *= 0.5f * __cosf(a2 * v + b2) + 0.5f;
      out[(size_t)i * Dc + j] = v;
    }
  }
}

// ---------------------------------------------------------------------------
extern "C" void kernel_launch(void* const* d_in, const int* in_sizes, int n_in,
                              void* d_out, int out_size, void* d_ws,
                              size_t ws_size, hipStream_t stream) {
  const float* x   = (const float*)d_in[0];
  const float* qw  = (const float*)d_in[1];
  const float* kw  = (const float*)d_in[2];
  const float* vw  = (const float*)d_in[3];
  const float* A1  = (const float*)d_in[4];
  const float* B1v = (const float*)d_in[5];
  const float* pw  = (const float*)d_in[6];
  const float* pbv = (const float*)d_in[7];
  const float* A2  = (const float*)d_in[8];
  const float* B2v = (const float*)d_in[9];
  float* out = (float*)d_out;

  char* base = (char*)d_ws;
  size_t off = 0;
  auto wsalloc = [&](size_t bytes) -> void* {
    void* p = base + off;
    off += (bytes + 255) & ~(size_t)255;
    return p;
  };
  bf16_t* xb  = (bf16_t*)wsalloc((size_t)Bc * Tc * Dc * 2);
  bf16_t* wqT = (bf16_t*)wsalloc((size_t)Hc * Sc * Dc * 2);
  bf16_t* wkT = (bf16_t*)wsalloc((size_t)Hc * Sc * Dc * 2);
  bf16_t* wvT = (bf16_t*)wsalloc((size_t)Hc * Sc * Dc * 2);
  bf16_t* pwb = (bf16_t*)wsalloc((size_t)Dc * Dc * 2);
  bf16_t* Qb  = (bf16_t*)wsalloc((size_t)Bc * Hc * Tc * Sc * 2);
  bf16_t* Kb  = (bf16_t*)wsalloc((size_t)Bc * Hc * Tc * Sc * 2);
  bf16_t* Vt  = (bf16_t*)wsalloc((size_t)Bc * Hc * Sc * Tc * 2);
  bf16_t* AO  = (bf16_t*)wsalloc((size_t)Bc * Tc * Dc * 2);

  // 1) precision conversion (+ weight transposes)
  conv_f2bf_kernel<<<(Bc * Tc * Dc) / 256, 256, 0, stream>>>(x, xb, Bc * Tc * Dc);
  conv_wT_kernel<<<(Hc * Sc * Dc) / 256, 256, 0, stream>>>(qw, wqT);
  conv_wT_kernel<<<(Hc * Sc * Dc) / 256, 256, 0, stream>>>(kw, wkT);
  conv_wT_kernel<<<(Hc * Sc * Dc) / 256, 256, 0, stream>>>(vw, wvT);
  conv_f2bf_kernel<<<(Dc * Dc) / 256, 256, 0, stream>>>(pw, pwb, Dc * Dc);

  // 2) QKV projections (WMMA + TDM-staged weights)
  qkv_kernel<<<(3 * Bc * Hc * (Tc / 16)) / 4, 128, 0, stream>>>(
      xb, wqT, wkT, wvT, Qb, Kb, Vt);

  // 3) two-pass flash attention + cos-dropout (WMMA)
  attn_kernel<<<(Bc * Hc * (Tc / 16)) / 4, 128, 0, stream>>>(
      Qb, Kb, Vt, A1, B1v, AO);

  // 4) output projection + bias + cos-dropout (WMMA + TDM-staged weights)
  proj_kernel<<<((Bc * Tc / 16) * (Dc / 64)) / 4, 128, 0, stream>>>(
      AO, pwb, pbv, A2, B2v, out);
}
